// MHA_60318520705419
// MI455X (gfx1250) — matrix operand order repaired and checked by reading, not compiled, not run on hardware
//
#include <hip/hip_runtime.h>
#include <hip/hip_bf16.h>

// ---------------------------------------------------------------------------
// MHA (B=2, S=2048, D=2560, H=32, HD=80, RD=32) for gfx1250 (MI455X, wave32).
// All matmuls run through v_wmma_f32_16x16x32_bf16 (fp32 accumulate).
// Flash row-sums ride on an extra WMMA (P x ones). K tiles staged with
// global_load_async_to_lds_b32 (ASYNCcnt path).
// ---------------------------------------------------------------------------

typedef __bf16 bf16_t;
typedef bf16_t v16bf __attribute__((ext_vector_type(16)));
typedef float  v8f   __attribute__((ext_vector_type(8)));

union Frag { v16bf v; unsigned int w[8]; };

__device__ __forceinline__ unsigned short f2bf(float f) {
  unsigned int u = __float_as_uint(f);
  u += 0x7FFFu + ((u >> 16) & 1u);           // round-to-nearest-even
  return (unsigned short)(u >> 16);
}
__device__ __forceinline__ float bf2f(unsigned short h) {
  return __uint_as_float(((unsigned int)h) << 16);
}

// ---------------------------------------------------------------------------
// GEMM: Out[M,N] = A[M,K] * W[N,K]^T + bias[N]
// Block: 256 threads (8 waves), tile 128(M) x 128(N), K-step 32.
// Wave grid 4(M) x 2(N); each wave owns 32x64 = 2x4 WMMA 16x16 tiles
// (8 WMMAs per K-step from 6 fragment loads).
// Fragment layouts follow CDNA5 ISA 7.12.2 (bf16 A 16x32, B 32x16, C 16x16).
// ---------------------------------------------------------------------------
template<bool A_IS_F32, bool OUT_IS_F32>
__global__ __launch_bounds__(256) void gemm_bias(
    const void* __restrict__ Av, const float* __restrict__ W,
    const float* __restrict__ bias, void* __restrict__ Outv,
    int M, int N, int K)
{
  __shared__ unsigned int a_sm32[128 * 16];   // 128 rows x 32 bf16
  __shared__ unsigned int b_sm32[128 * 16];   // 128 rows x 32 bf16
  unsigned short* a_sm = (unsigned short*)a_sm32;
  unsigned short* b_sm = (unsigned short*)b_sm32;

  const int tid  = threadIdx.x;
  const int lane = tid & 31, wid = tid >> 5;
  const int wm = wid & 3, wn = wid >> 2;
  const int m = lane & 15, hi = lane >> 4;

  const int row0 = blockIdx.y * 128;
  const int col0 = blockIdx.x * 128;

  const v8f vzero = {0.f, 0.f, 0.f, 0.f, 0.f, 0.f, 0.f, 0.f};
  v8f acc[2][4];
  #pragma unroll
  for (int i = 0; i < 2; ++i)
    #pragma unroll
    for (int j = 0; j < 4; ++j) acc[i][j] = vzero;

  const int lr = tid & 127, lc = (tid >> 7) * 16;   // tile loaders: row, col-base

  for (int kb = 0; kb < K; kb += 32) {
    // ---- stage A tile (convert f32 -> bf16 if needed) ----
    if constexpr (A_IS_F32) {
      const float* A = (const float*)Av;
      const float4* src = (const float4*)&A[(size_t)(row0 + lr) * K + kb + lc];
      #pragma unroll
      for (int q = 0; q < 4; ++q) {
        float4 f = src[q];
        int base = lr * 32 + lc + q * 4;
        a_sm[base + 0] = f2bf(f.x); a_sm[base + 1] = f2bf(f.y);
        a_sm[base + 2] = f2bf(f.z); a_sm[base + 3] = f2bf(f.w);
      }
      if (kb + 32 < K)
        __builtin_prefetch(&A[(size_t)(row0 + lr) * K + kb + 32 + lc], 0, 1);
    } else {
      const unsigned short* A = (const unsigned short*)Av;
      const uint4* src = (const uint4*)&A[(size_t)(row0 + lr) * K + kb + lc];
      uint4* dst = (uint4*)&a_sm[lr * 32 + lc];
      dst[0] = src[0]; dst[1] = src[1];
      if (kb + 32 < K)
        __builtin_prefetch(&A[(size_t)(row0 + lr) * K + kb + 32 + lc], 0, 1);
    }
    // ---- stage B tile from W (f32 -> bf16) ----
    {
      const float4* src = (const float4*)&W[(size_t)(col0 + lr) * K + kb + lc];
      #pragma unroll
      for (int q = 0; q < 4; ++q) {
        float4 f = src[q];
        int base = lr * 32 + lc + q * 4;
        b_sm[base + 0] = f2bf(f.x); b_sm[base + 1] = f2bf(f.y);
        b_sm[base + 2] = f2bf(f.z); b_sm[base + 3] = f2bf(f.w);
      }
      if (kb + 32 < K)
        __builtin_prefetch(&W[(size_t)(col0 + lr) * K + kb + 32 + lc], 0, 1);
    }
    __syncthreads();

    Frag afr[2], bfr[4];
    #pragma unroll
    for (int i = 0; i < 2; ++i) {
      int arow = wm * 32 + i * 16 + m;
      #pragma unroll
      for (int v = 0; v < 8; ++v) {
        int k = ((v < 4) ? 0 : 16) + hi * 8 + (v & 3) * 2;   // ISA A-layout
        afr[i].w[v] = a_sm32[arow * 16 + (k >> 1)];
      }
    }
    #pragma unroll
    for (int j = 0; j < 4; ++j) {
      int brow = wn * 64 + j * 16 + m;
      #pragma unroll
      for (int v = 0; v < 8; ++v) {
        int k = ((v < 4) ? 0 : 16) + hi * 8 + (v & 3) * 2;   // ISA B-layout
        bfr[j].w[v] = b_sm32[brow * 16 + (k >> 1)];
      }
    }
    #pragma unroll
    for (int i = 0; i < 2; ++i)
      #pragma unroll
      for (int j = 0; j < 4; ++j)
        acc[i][j] = __builtin_amdgcn_wmma_f32_16x16x32_bf16(
            false, afr[i].v, false, bfr[j].v, (short)0, acc[i][j], false, false);
    __syncthreads();
  }

  // ---- epilogue: bias add + store ----
  #pragma unroll
  for (int i = 0; i < 2; ++i) {
    #pragma unroll
    for (int j = 0; j < 4; ++j) {
      int colg = col0 + wn * 64 + j * 16 + m;
      float bv = bias[colg];
      #pragma unroll
      for (int r = 0; r < 8; ++r) {
        int rowg = row0 + wm * 32 + i * 16 + r + 8 * hi;     // ISA C-layout
        float val = acc[i][j][r] + bv;
        if constexpr (OUT_IS_F32)
          ((float*)Outv)[(size_t)rowg * N + colg] = val;
        else
          ((unsigned short*)Outv)[(size_t)rowg * N + colg] = f2bf(val);
      }
    }
  }
}

// ---------------------------------------------------------------------------
// RoPE, in place on bf16 qkv [B,S,3,H,80]; rotary pairs (i, i+16), i<16.
// ---------------------------------------------------------------------------
__global__ __launch_bounds__(256) void rope_kernel(unsigned short* __restrict__ qkv)
{
  int tid = blockIdx.x * blockDim.x + threadIdx.x;  // 2^22 threads
  int i  = tid & 15;
  int h  = (tid >> 4) & 31;
  int t3 = (tid >> 9) & 1;                          // 0 = q, 1 = k
  int s  = (tid >> 10) & 2047;
  int b  = tid >> 21;
  size_t base = ((size_t)(b * 2048 + s) * 3 + t3) * 2560 + (size_t)h * 80;
  float inv = __powf(10000.f, -(float)i / 16.f);
  float sn, cs;
  __sincosf((float)s * inv, &sn, &cs);
  float t1 = bf2f(qkv[base + i]);
  float t2 = bf2f(qkv[base + i + 16]);
  qkv[base + i]      = f2bf(t1 * cs - t2 * sn);
  qkv[base + i + 16] = f2bf(t1 * sn + t2 * cs);
}

// ---------------------------------------------------------------------------
// Causal flash attention. Grid (S/128, H, B); 8 waves, 16 queries per wave.
// Per 32-key tile: QK^T = 2x3 WMMAs (HD 80 padded to 96, A zero-padded),
// online softmax (shfl_xor row-max only; row-sum computed by an extra
// WMMA against an all-ones B fragment), P re-laid out via LDS, then
// P*V = 5 WMMAs into fp32 accumulators. K tile staged with
// global_load_async_to_lds_b32. ctx written as bf16 [B,S,H*80].
// ---------------------------------------------------------------------------
__global__ __launch_bounds__(256) void flash_attn(
    const unsigned short* __restrict__ qkv, unsigned short* __restrict__ ctx)
{
  __shared__ unsigned int k_sm32[32 * 40];          // K tile 32 keys x 80 bf16
  __shared__ unsigned int vt_sm32[80 * 18];         // V^T tile 80 hd x 32 keys (+pad)
  __shared__ unsigned int p_sm32[8 * 16 * 16];      // per-wave P 16x32 bf16
  unsigned short* vt_sm = (unsigned short*)vt_sm32;
  unsigned short* p_sm  = (unsigned short*)p_sm32;

  const int tid = threadIdx.x, lane = tid & 31, wid = tid >> 5;
  const int m = lane & 15, hi = lane >> 4;
  const int qbase = blockIdx.x * 128;
  const int h = blockIdx.y, b = blockIdx.z;
  const int qw = qbase + wid * 16;                  // this wave's first query

  // Q fragments: rows qw..qw+15, K-dim padded 80 -> 96 with zeros
  Frag qf[3];
  {
    const unsigned int* qrow = (const unsigned int*)(qkv +
        ((size_t)(b * 2048 + qw + m) * 3 + 0) * 2560 + (size_t)h * 80);
    #pragma unroll
    for (int c = 0; c < 3; ++c)
      #pragma unroll
      for (int v = 0; v < 8; ++v) {
        int hd = c * 32 + ((v < 4) ? 0 : 16) + hi * 8 + (v & 3) * 2;
        qf[c].w[v] = (hd < 80) ? qrow[hd >> 1] : 0u;
      }
  }

  // all-ones bf16 B fragment (for row sums via the matrix unit)
  Frag ones;
  #pragma unroll
  for (int v = 0; v < 8; ++v) ones.w[v] = 0x3F803F80u;

  const v8f vzero = {0.f, 0.f, 0.f, 0.f, 0.f, 0.f, 0.f, 0.f};
  v8f acc[6];                                       // [0..4]=ctx hd tiles, [5]=row sum
  #pragma unroll
  for (int t = 0; t < 6; ++t) acc[t] = vzero;
  float mrow[8];
  #pragma unroll
  for (int r = 0; r < 8; ++r) mrow[r] = -1e30f;

  const float scale = 0.11180339887498948f;         // 1/sqrt(80)
  const int nkt = (qbase + 128) >> 5;
  const int key_ld = tid >> 3, seg = tid & 7;       // cooperative tile loaders

  for (int kt = 0; kt < nkt; ++kt) {
    const int kb = kt << 5;
    {   // K tile, row-major: async global -> LDS DMA (ASYNCcnt)
      const unsigned short* krow = qkv +
          ((size_t)(b * 2048 + kb + key_ld) * 3 + 1) * 2560 + (size_t)h * 80;
      unsigned lds_off = (unsigned)(size_t)&k_sm32[key_ld * 40 + seg * 5];
      #pragma unroll
      for (int e = 0; e < 5; ++e)
        asm volatile("global_load_async_to_lds_b32 %0, %1, off"
                     :: "v"(lds_off + 4u * e), "v"(krow + 2 * (seg * 5 + e))
                     : "memory");
    }
    {   // V tile, transposed into vt_sm[hd][key]
      const unsigned int* vrow = (const unsigned int*)(qkv +
          ((size_t)(b * 2048 + kb + key_ld) * 3 + 2) * 2560 + (size_t)h * 80);
      #pragma unroll
      for (int e = 0; e < 5; ++e) {
        unsigned int u = vrow[seg * 5 + e];
        int hd = seg * 10 + e * 2;
        vt_sm[hd * 36 + key_ld]       = (unsigned short)(u & 0xFFFFu);
        vt_sm[(hd + 1) * 36 + key_ld] = (unsigned short)(u >> 16);
      }
    }
    asm volatile("s_wait_asynccnt 0" ::: "memory"); // K async DMA done
    __syncthreads();

    if (kb <= qw + 15) {                            // causal: wave participates
      // ---- scores: S = Q * K^T ----
      v8f sc[2]; sc[0] = vzero; sc[1] = vzero;
      #pragma unroll
      for (int j = 0; j < 2; ++j) {
        #pragma unroll
        for (int c = 0; c < 3; ++c) {
          Frag bf;
          #pragma unroll
          for (int v = 0; v < 8; ++v) {
            int k = c * 32 + ((v < 4) ? 0 : 16) + hi * 8 + (v & 3) * 2;
            bf.w[v] = (k < 80) ? k_sm32[(j * 16 + m) * 40 + (k >> 1)] : 0u;
          }
          sc[j] = __builtin_amdgcn_wmma_f32_16x16x32_bf16(
              false, qf[c].v, false, bf.v, (short)0, sc[j], false, false);
        }
      }
      // ---- mask + scale ----
      float sv[2][8];
      #pragma unroll
      for (int j = 0; j < 2; ++j) {
        int keyg = kb + j * 16 + m;
        #pragma unroll
        for (int r = 0; r < 8; ++r) {
          int qg = qw + r + 8 * hi;
          float v = sc[j][r] * scale;
          sv[j][r] = (keyg > qg) ? -1e30f : v;
        }
      }
      // ---- online softmax: row-max butterflies only ----
      #pragma unroll
      for (int r = 0; r < 8; ++r) {
        float mx = fmaxf(sv[0][r], sv[1][r]);
        #pragma unroll
        for (int off = 1; off < 16; off <<= 1)
          mx = fmaxf(mx, __shfl_xor(mx, off, 32));
        float mnew  = fmaxf(mrow[r], mx);
        float alpha = __expf(mrow[r] - mnew);
        float p0 = __expf(sv[0][r] - mnew);
        float p1 = __expf(sv[1][r] - mnew);
        p_sm[wid * 512 + (r + 8 * hi) * 32 + m]      = f2bf(p0);
        p_sm[wid * 512 + (r + 8 * hi) * 32 + 16 + m] = f2bf(p1);
        mrow[r] = mnew;
        #pragma unroll
        for (int t = 0; t < 6; ++t) acc[t][r] *= alpha;
      }
      asm volatile("s_wait_dscnt 0" ::: "memory");  // per-wave P write -> read

      // ---- ctx += P * V ; rowsum += P * ones ----
      Frag pf;
      #pragma unroll
      for (int v = 0; v < 8; ++v) {
        int k = ((v < 4) ? 0 : 16) + hi * 8 + (v & 3) * 2;
        pf.w[v] = p_sm32[wid * 256 + m * 16 + (k >> 1)];
      }
      #pragma unroll
      for (int t = 0; t < 5; ++t) {
        Frag bf;
        #pragma unroll
        for (int v = 0; v < 8; ++v) {
          int kk = ((v < 4) ? 0 : 16) + hi * 8 + (v & 3) * 2;
          bf.w[v] = vt_sm32[(t * 16 + m) * 18 + (kk >> 1)];
        }
        acc[t] = __builtin_amdgcn_wmma_f32_16x16x32_bf16(
            false, pf.v, false, bf.v, (short)0, acc[t], false, false);
      }
      acc[5] = __builtin_amdgcn_wmma_f32_16x16x32_bf16(
          false, pf.v, false, ones.v, (short)0, acc[5], false, false);
    }
    __syncthreads();
  }

  // ---- normalize + store ctx (bf16); acc[5] holds l replicated per lane ----
  #pragma unroll
  for (int t = 0; t < 5; ++t) {
    #pragma unroll
    for (int r = 0; r < 8; ++r) {
      float val = acc[t][r] / acc[5][r];
      size_t idx = (size_t)(b * 2048 + qw + r + 8 * hi) * 2560
                 + (size_t)h * 80 + t * 16 + m;
      ctx[idx] = f2bf(val);
    }
  }
}

// ---------------------------------------------------------------------------
extern "C" void kernel_launch(void* const* d_in, const int* in_sizes, int n_in,
                              void* d_out, int out_size, void* d_ws, size_t ws_size,
                              hipStream_t stream) {
  const float* x      = (const float*)d_in[0];   // [2,2048,2560]
  const float* wqkv_w = (const float*)d_in[1];   // [7680,2560]
  const float* wqkv_b = (const float*)d_in[2];   // [7680]
  const float* out_w  = (const float*)d_in[3];   // [2560,2560]
  const float* out_b  = (const float*)d_in[4];   // [2560]
  float* out = (float*)d_out;                    // [2,2048,2560]

  // workspace: qkv bf16 (4096*7680 = 60MB) + ctx bf16 (4096*2560 = 20MB)
  unsigned short* qkv = (unsigned short*)d_ws;
  unsigned short* ctx = qkv + (size_t)4096 * 7680;

  // 1) qkv = x @ wqkv_w^T + wqkv_b   (M=4096, N=7680, K=2560)
  gemm_bias<true, false><<<dim3(7680 / 128, 4096 / 128), 256, 0, stream>>>(
      x, wqkv_w, wqkv_b, qkv, 4096, 7680, 2560);

  // 2) RoPE on q,k halves (in place)
  rope_kernel<<<16384, 256, 0, stream>>>(qkv);

  // 3) causal flash attention -> ctx
  flash_attn<<<dim3(16, 32, 2), 256, 0, stream>>>(qkv, ctx);

  // 4) out = ctx @ out_w^T + out_b   (M=4096, N=2560, K=2560)
  gemm_bias<false, true><<<dim3(2560 / 128, 4096 / 128), 256, 0, stream>>>(
      ctx, out_w, out_b, out, 4096, 2560, 2560);
}
